// Attention_51599737094204
// MI455X (gfx1250) — matrix operand order repaired and checked by reading, not compiled, not run on hardware
//
#include <hip/hip_runtime.h>

#define DIMM   768
#define HEADS  12
#define DHEAD  64
#define INNER  768
#define BATCH  8
#define SEQ    1024
#define SCALEF 0.125f  // 64^-0.5

typedef __attribute__((ext_vector_type(16))) __bf16 v16bf;
typedef __attribute__((ext_vector_type(8)))  __bf16 v8bf;
typedef __attribute__((ext_vector_type(8)))  float  v8f;
typedef __attribute__((ext_vector_type(4)))  float  v4f;

__device__ __forceinline__ v8f wmma_bf16(v16bf a, v16bf b, v8f c) {
  // D = A(16x32 bf16) * B(32x16 bf16) + C(16x16 f32)
  return __builtin_amdgcn_wmma_f32_16x16x32_bf16(false, a, false, b, (short)0, c,
                                                 false, false);
}

__device__ __forceinline__ v16bf cat8(v8bf lo, v8bf hi) {
  return __builtin_shufflevector(lo, hi, 0, 1, 2, 3, 4, 5, 6, 7, 8, 9, 10, 11,
                                 12, 13, 14, 15);
}

// ---- fragment loaders (CDNA5 wave32 layouts, ISA 7.12.2) ----
// A 16x32 bf16: lane L(<16) = row L, K = 0..7 & 16..23 ; lane L+16 = row L, K = 8..15 & 24..31
// All callers guarantee 16-byte alignment of each 4-float / 8-bf16 chunk.
__device__ __forceinline__ v16bf load_a_f32(const float* __restrict__ src, int ld) {
  const int lane = threadIdx.x & 31;
  const int row  = lane & 15;
  const int kh   = (lane >> 4) << 3;  // 0 or 8
  const float* p = src + (size_t)row * ld + kh;
  v4f f0 = *(const v4f*)(p + 0);
  v4f f1 = *(const v4f*)(p + 4);
  v4f f2 = *(const v4f*)(p + 16);
  v4f f3 = *(const v4f*)(p + 20);
  v16bf a;
#pragma unroll
  for (int e = 0; e < 4; ++e) {
    a[e]      = (__bf16)f0[e];
    a[4 + e]  = (__bf16)f1[e];
    a[8 + e]  = (__bf16)f2[e];
    a[12 + e] = (__bf16)f3[e];
  }
  return a;
}
__device__ __forceinline__ v16bf load_a_bf16(const __bf16* __restrict__ src, int ld) {
  const int lane = threadIdx.x & 31;
  const int row  = lane & 15;
  const int kh   = (lane >> 4) << 3;
  const __bf16* p = src + (size_t)row * ld + kh;
  v8bf lo = *(const v8bf*)(p);
  v8bf hi = *(const v8bf*)(p + 16);
  return cat8(lo, hi);
}
// B 32x16 bf16 (row-major KxN): lane L(<16) = col L, K = 0..7 & 16..23 ; lane L+16 = col L, K = 8..15 & 24..31
__device__ __forceinline__ v16bf load_b_f32(const float* __restrict__ src, int ld) {
  const int lane = threadIdx.x & 31;
  const float* p = src + (size_t)((lane >> 4) << 3) * ld + (lane & 15);
  v16bf b;
#pragma unroll
  for (int s = 0; s < 8; ++s) {
    b[s]     = (__bf16)p[(size_t)s * ld];
    b[8 + s] = (__bf16)p[(size_t)(16 + s) * ld];
  }
  return b;
}
__device__ __forceinline__ v16bf load_b_bf16(const __bf16* __restrict__ src, int ld) {
  const int lane = threadIdx.x & 31;
  const __bf16* p = src + (size_t)((lane >> 4) << 3) * ld + (lane & 15);
  v16bf b;
#pragma unroll
  for (int s = 0; s < 8; ++s) {
    b[s]     = p[(size_t)s * ld];
    b[8 + s] = p[(size_t)(16 + s) * ld];
  }
  return b;
}

// ============================================================================
// Kernel 1: fused QKV projection. x(8192,768)f32 @ [Wq|Wkv](768,2304)f32
// -> Q (b,h,n,d) bf16 ; Kt (b,h,d,n) bf16 (transposed!) ; V (b,h,n,d) bf16
// one wave computes a 16x64 output tile; software-pipelined K-loop of 24 steps.
// ============================================================================
__global__ __launch_bounds__(256) void qkv_proj_kernel(
    const float* __restrict__ x, const float* __restrict__ wq,
    const float* __restrict__ wkv, __bf16* __restrict__ Q,
    __bf16* __restrict__ Kt, __bf16* __restrict__ V) {
  const int wid  = blockIdx.x * (blockDim.x >> 5) + (threadIdx.x >> 5);
  const int NT64 = (3 * INNER) / 64;  // 36
  const int mt   = wid / NT64;        // 0..511
  const int nt   = wid % NT64;
  const int m0   = mt * 16;
  const int n0   = nt * 64;
  const int lane = threadIdx.x & 31;

  // hoist source-matrix selection out of the K loop (wave-invariant)
  const float* bbase;
  int ldb;
  if (n0 < INNER) { bbase = wq + n0; ldb = INNER; }
  else            { bbase = wkv + (n0 - INNER); ldb = 2 * INNER; }
  const float* abase = x + (size_t)m0 * DIMM;

  v8f acc[4] = {};
  // software pipeline: loads for step k+1 issue before WMMAs of step k
  v16bf a = load_a_f32(abase, DIMM);
  v16bf b[4];
#pragma unroll
  for (int t = 0; t < 4; ++t) b[t] = load_b_f32(bbase + t * 16, ldb);

  for (int kk = 0; kk < DIMM; kk += 32) {
    const int kn = (kk + 32 < DIMM) ? kk + 32 : 0;  // clamped (redundant last load)
    v16bf an = load_a_f32(abase + kn, DIMM);
    v16bf bn[4];
#pragma unroll
    for (int t = 0; t < 4; ++t)
      bn[t] = load_b_f32(bbase + (size_t)kn * ldb + t * 16, ldb);
#pragma unroll
    for (int t = 0; t < 4; ++t) acc[t] = wmma_bf16(a, b[t], acc[t]);
    a = an;
#pragma unroll
    for (int t = 0; t < 4; ++t) b[t] = bn[t];
  }

  // scatter-store into per-head layouts (C frag: row = r + 8*(lane/16), col = lane%16)
#pragma unroll
  for (int t = 0; t < 4; ++t) {
    const int c = n0 + t * 16 + (lane & 15);
#pragma unroll
    for (int r = 0; r < 8; ++r) {
      const int m  = m0 + r + ((lane >> 4) << 3);
      const int bb = m >> 10, i = m & 1023;
      const float v = acc[t][r];
      if (c < INNER) {
        const int h = c >> 6, d = c & 63;
        Q[((size_t)(bb * HEADS + h) * SEQ + i) * DHEAD + d] = (__bf16)v;
      } else if (c < 2 * INNER) {
        const int ck = c - INNER, h = ck >> 6, d = ck & 63;
        Kt[((size_t)(bb * HEADS + h) * DHEAD + d) * SEQ + i] = (__bf16)v;  // transposed
      } else {
        const int cv = c - 2 * INNER, h = cv >> 6, d = cv & 63;
        V[((size_t)(bb * HEADS + h) * SEQ + i) * DHEAD + d] = (__bf16)v;
      }
    }
  }
}

// ============================================================================
// Kernel 2: fused talking-heads attention. 12 waves/WG (wave g <-> head g),
// 16 queries per WG, two-pass exact softmax, all score tiles stay in LDS.
// LDS rows padded to keep 16B alignment for ds_load/store_b128.
// ============================================================================
#define SCPAD 36  // 144B rows, 16B aligned
#define ATPAD 40  // 80B rows, 16B aligned

__global__ __launch_bounds__(384) void talking_heads_attn_kernel(
    const __bf16* __restrict__ Q, const __bf16* __restrict__ Kt,
    const __bf16* __restrict__ V, const float* __restrict__ mix_pre,
    const float* __restrict__ mix_post, float* __restrict__ att_out) {
  __shared__ float  s_sc[HEADS][16][SCPAD];  // raw scores S_h (16 q x 32 j)
  __shared__ __bf16 s_at[HEADS][16][ATPAD];  // normalized attn_h
  __shared__ __bf16 s_pa[HEADS][16][ATPAD];  // post-mixed probs A_g
  __shared__ float  s_m[HEADS][16], s_il[HEADS][16];
  __shared__ float  s_mpre[HEADS][HEADS], s_mpost[HEADS][HEADS];

  const int b    = blockIdx.x >> 6;
  const int i0   = (blockIdx.x & 63) << 4;
  const int g    = threadIdx.x >> 5;  // wave id == head id
  const int lane = threadIdx.x & 31;

  if (threadIdx.x < HEADS * HEADS) {
    // fold the q*k scale into the pre-softmax mix (linear)
    s_mpre[threadIdx.x / HEADS][threadIdx.x % HEADS]  = mix_pre[threadIdx.x] * SCALEF;
    s_mpost[threadIdx.x / HEADS][threadIdx.x % HEADS] = mix_post[threadIdx.x];
  }
  __syncthreads();

  const __bf16* qbase  = Q + ((size_t)(b * HEADS + g) * SEQ + i0) * DHEAD;
  const __bf16* ktbase = Kt + (size_t)(b * HEADS + g) * DHEAD * SEQ;
  const __bf16* vbase  = V + (size_t)(b * HEADS + g) * SEQ * DHEAD;

  const v16bf qa0 = load_a_bf16(qbase, DHEAD);       // K(d) = 0..31
  const v16bf qa1 = load_a_bf16(qbase + 32, DHEAD);  // K(d) = 32..63

  const int srow  = lane >> 1;        // each lane: one query row, half of j-columns
  const int cbase = (lane & 1) << 4;
  float m_run = -3.0e38f, l_run = 0.f;

  // -------- pass 1: per-head running (max, sum) of mixed scores --------
  for (int j0 = 0; j0 < SEQ; j0 += 32) {
#pragma unroll
    for (int sub = 0; sub < 2; ++sub) {
      const int j = j0 + sub * 16;
      v16bf b0 = load_b_bf16(ktbase + j, SEQ);             // d rows 0..31
      v16bf b1 = load_b_bf16(ktbase + 32 * SEQ + j, SEQ);  // d rows 32..63
      v8f s = {};
      s = wmma_bf16(qa0, b0, s);
      s = wmma_bf16(qa1, b1, s);
#pragma unroll
      for (int r = 0; r < 8; ++r)
        s_sc[g][r + ((lane >> 4) << 3)][sub * 16 + (lane & 15)] = s[r];
    }
    __syncthreads();
    v4f t4[4] = {};
    for (int hh = 0; hh < HEADS; ++hh) {
      const float w = s_mpre[hh][g];
      const v4f* pr = (const v4f*)&s_sc[hh][srow][cbase];  // b128 LDS loads
#pragma unroll
      for (int q = 0; q < 4; ++q) t4[q] += pr[q] * w;
    }
    float bm = -3.0e38f;
#pragma unroll
    for (int q = 0; q < 4; ++q)
#pragma unroll
      for (int e = 0; e < 4; ++e) bm = fmaxf(bm, t4[q][e]);
    const float mnew = fmaxf(m_run, bm);
    float lsum = 0.f;
#pragma unroll
    for (int q = 0; q < 4; ++q)
#pragma unroll
      for (int e = 0; e < 4; ++e) lsum += __expf(t4[q][e] - mnew);
    l_run = l_run * __expf(m_run - mnew) + lsum;
    m_run = mnew;
    __syncthreads();
  }
  {
    const float om = __shfl_xor(m_run, 1, 32);
    const float ol = __shfl_xor(l_run, 1, 32);
    const float M  = fmaxf(m_run, om);
    const float L  = l_run * __expf(m_run - M) + ol * __expf(om - M);
    if ((lane & 1) == 0) { s_m[g][srow] = M; s_il[g][srow] = 1.0f / L; }
  }
  __syncthreads();

  // -------- pass 2: normalize, post-mix across heads, accumulate P@V --------
  v8f o[4] = {};
  for (int j0 = 0; j0 < SEQ; j0 += 32) {
#pragma unroll
    for (int sub = 0; sub < 2; ++sub) {
      const int j = j0 + sub * 16;
      v16bf b0 = load_b_bf16(ktbase + j, SEQ);
      v16bf b1 = load_b_bf16(ktbase + 32 * SEQ + j, SEQ);
      v8f s = {};
      s = wmma_bf16(qa0, b0, s);
      s = wmma_bf16(qa1, b1, s);
#pragma unroll
      for (int r = 0; r < 8; ++r)
        s_sc[g][r + ((lane >> 4) << 3)][sub * 16 + (lane & 15)] = s[r];
    }
    __syncthreads();
    {  // this head's normalized softmax tile
      const float m  = s_m[g][srow];
      const float il = s_il[g][srow];
      v4f t4[4] = {};
      for (int hh = 0; hh < HEADS; ++hh) {
        const float w = s_mpre[hh][g];
        const v4f* pr = (const v4f*)&s_sc[hh][srow][cbase];
#pragma unroll
        for (int q = 0; q < 4; ++q) t4[q] += pr[q] * w;
      }
      v8bf av[2];
#pragma unroll
      for (int q = 0; q < 4; ++q)
#pragma unroll
        for (int e = 0; e < 4; ++e)
          av[q >> 1][(q & 1) * 4 + e] = (__bf16)(__expf(t4[q][e] - m) * il);
      v8bf* dst = (v8bf*)&s_at[g][srow][cbase];  // b128 LDS stores
      dst[0] = av[0];
      dst[1] = av[1];
    }
    __syncthreads();
    {  // post-softmax talking-heads mix
      v4f t4[4] = {};
      for (int hh = 0; hh < HEADS; ++hh) {
        const float w = s_mpost[hh][g];
        const v8bf* pr = (const v8bf*)&s_at[hh][srow][cbase];
        v8bf a0 = pr[0], a1 = pr[1];
#pragma unroll
        for (int e = 0; e < 8; ++e) {
          t4[e >> 2][e & 3]       += w * (float)a0[e];
          t4[2 + (e >> 2)][e & 3] += w * (float)a1[e];
        }
      }
      v8bf pv[2];
#pragma unroll
      for (int q = 0; q < 4; ++q)
#pragma unroll
        for (int e = 0; e < 4; ++e) pv[q >> 1][(q & 1) * 4 + e] = (__bf16)t4[q][e];
      v8bf* dst = (v8bf*)&s_pa[g][srow][cbase];
      dst[0] = pv[0];
      dst[1] = pv[1];
    }
    __syncthreads();
    // A-fragment of mixed probs (16x32), then P @ V (4 WMMAs, K=32)
    v16bf pa;
    {
      const int row = lane & 15;
      const int kh  = (lane >> 4) << 3;
      v8bf lo = *(const v8bf*)&s_pa[g][row][kh];
      v8bf hi = *(const v8bf*)&s_pa[g][row][16 + kh];
      pa = cat8(lo, hi);
    }
#pragma unroll
    for (int t = 0; t < 4; ++t) {
      v16bf vb = load_b_bf16(vbase + (size_t)j0 * DHEAD + t * 16, DHEAD);
      o[t] = wmma_bf16(pa, vb, o[t]);
    }
  }

  // store O as f32 (b, i, g*64+d) for the output projection
#pragma unroll
  for (int t = 0; t < 4; ++t) {
    const int d = t * 16 + (lane & 15);
#pragma unroll
    for (int r = 0; r < 8; ++r) {
      const int i = i0 + r + ((lane >> 4) << 3);
      att_out[((size_t)b * SEQ + i) * INNER + g * DHEAD + d] = o[t][r];
    }
  }
}

// ============================================================================
// Kernel 3: output projection  att(8192,768) @ W_out(768,768) + b_out
// ============================================================================
__global__ __launch_bounds__(256) void out_proj_kernel(
    const float* __restrict__ att, const float* __restrict__ wout,
    const float* __restrict__ bout, float* __restrict__ out) {
  const int wid  = blockIdx.x * (blockDim.x >> 5) + (threadIdx.x >> 5);
  const int NT64 = DIMM / 64;  // 12
  const int mt   = wid / NT64;
  const int nt   = wid % NT64;
  const int m0   = mt * 16, n0 = nt * 64;
  const int lane = threadIdx.x & 31;

  const float* abase = att + (size_t)m0 * INNER;
  const float* bbase = wout + n0;

  v8f acc[4] = {};
  v16bf a = load_a_f32(abase, INNER);
  v16bf b[4];
#pragma unroll
  for (int t = 0; t < 4; ++t) b[t] = load_b_f32(bbase + t * 16, DIMM);

  for (int kk = 0; kk < INNER; kk += 32) {
    const int kn = (kk + 32 < INNER) ? kk + 32 : 0;
    v16bf an = load_a_f32(abase + kn, INNER);
    v16bf bn[4];
#pragma unroll
    for (int t = 0; t < 4; ++t)
      bn[t] = load_b_f32(bbase + (size_t)kn * DIMM + t * 16, DIMM);
#pragma unroll
    for (int t = 0; t < 4; ++t) acc[t] = wmma_bf16(a, b[t], acc[t]);
    a = an;
#pragma unroll
    for (int t = 0; t < 4; ++t) b[t] = bn[t];
  }

#pragma unroll
  for (int t = 0; t < 4; ++t) {
    const int c     = n0 + t * 16 + (lane & 15);
    const float bia = bout[c];
#pragma unroll
    for (int r = 0; r < 8; ++r) {
      const int m = m0 + r + ((lane >> 4) << 3);
      out[(size_t)m * DIMM + c] = acc[t][r] + bia;
    }
  }
}

// ============================================================================
extern "C" void kernel_launch(void* const* d_in, const int* in_sizes, int n_in,
                              void* d_out, int out_size, void* d_ws, size_t ws_size,
                              hipStream_t stream) {
  const float* x        = (const float*)d_in[0];
  const float* W_q      = (const float*)d_in[1];
  const float* W_kv     = (const float*)d_in[2];
  const float* mix_pre  = (const float*)d_in[3];
  const float* mix_post = (const float*)d_in[4];
  const float* W_out    = (const float*)d_in[5];
  const float* b_out    = (const float*)d_in[6];
  float* out = (float*)d_out;
  (void)in_sizes; (void)n_in; (void)out_size; (void)ws_size;

  // workspace layout: Q | Kt | V (bf16) | att_out (f32)  ~= 63 MB
  const size_t nPerHead = (size_t)BATCH * HEADS * SEQ * DHEAD;  // 6,291,456
  __bf16* Qb  = (__bf16*)d_ws;
  __bf16* Ktb = Qb + nPerHead;
  __bf16* Vb  = Ktb + nPerHead;
  float*  att = (float*)(Vb + nPerHead);

  // 1) QKV projection: 512 M-tiles x 36 N64-tiles = 18432 waves / 8 per block
  qkv_proj_kernel<<<dim3(2304), dim3(256), 0, stream>>>(x, W_q, W_kv, Qb, Ktb, Vb);

  // 2) fused talking-heads attention: (8 batches x 64 query-tiles) x 12 waves
  talking_heads_attn_kernel<<<dim3(512), dim3(384), 0, stream>>>(
      Qb, Ktb, Vb, mix_pre, mix_post, att);

  // 3) output projection: 512 x 12 = 6144 waves / 8 per block
  out_proj_kernel<<<dim3(768), dim3(256), 0, stream>>>(att, W_out, b_out, out);
}
